// GCN1_52029233824243
// MI455X (gfx1250) — matrix operand order repaired and checked by reading, not compiled, not run on hardware
//
#include <hip/hip_runtime.h>
#include <hip/hip_bf16.h>

// ---------------------------------------------------------------------------
// Problem constants (from reference)
// ---------------------------------------------------------------------------
#define BG     32
#define NN     1024
#define DEGC   8
#define F_IN   512
#define HD     1024
#define OUTC   10
#define EE     (BG * NN * DEGC)   // 262144
#define KP1    (NN / 2)           // 512
#define KP2    (NN / 4)           // 256
#define N1     (BG * NN)          // 32768 nodes layer 1
#define N2     (BG * KP1)         // 16384 nodes layer 2
#define N3     (BG * KP2)         // 8192 nodes after pool 2
#define EPSBN  1e-5f

// ---------------------------------------------------------------------------
// WMMA types (CDNA5 gfx1250, wave32)
// ---------------------------------------------------------------------------
typedef __attribute__((ext_vector_type(16))) __bf16          v16bf;
typedef __attribute__((ext_vector_type(8)))  float           v8f;
typedef __attribute__((ext_vector_type(8)))  unsigned short  v8us;   // 16 bytes

union FragAB { v16bf v; v8us h[2]; unsigned short u[16]; };
union FragC  { v8f   v; float f[8]; };

__device__ __forceinline__ unsigned short f32_to_bf16(float f) {
  unsigned int u = __float_as_uint(f);
  u += 0x7FFFu + ((u >> 16) & 1u);   // round-to-nearest-even
  return (unsigned short)(u >> 16);
}

// ---------------------------------------------------------------------------
// WMMA GEMM:  C[M,N] = A[M,K](f32) * Bt[N,K](bf16, pre-transposed)  (+bias)(+relu)
//
// Block: 256 threads = 8 waves arranged 4(M) x 2(N) over a 128x128 tile.
// Each wave owns a 32x64 sub-tile: 2 A-frags x 4 B-frags = 8 WMMAs per K-step
// fed by 12 ds_load_b128 (all fragment loads issued before the WMMA burst).
// K-step 32, ping-pong LDS (2 x 20 KB of the WGP's 320 KB).
//
// LDS layouts give every fragment as contiguous, 16B-aligned chunks:
//   As[m][k]  row stride 40 u16 = 80B (mult of 16)
//   Bs[n][k]  row stride 40 u16 = 80B   (B pre-transposed to [N][K] in HBM)
// Fragment maps follow the ISA wave32 layouts:
//   A 16x32 (16-bit): lane lh holds M=lh; K = {hi*8+0..7, 16+hi*8+0..7}
//   B 32x16 (16-bit): lane lh holds N=lh; K = {hi*8+0..7, 16+hi*8+0..7}
//   C/D f32 16x16:    lane = (m>=8?16:0)+n; VGPR r = m%8
// ---------------------------------------------------------------------------
#define BM 128
#define BN 128
#define BK 32
#define LDP (BK + 8)   // 40 u16 = 80 bytes row stride

__device__ __forceinline__ void stage_tiles(
    unsigned short (*As)[LDP], unsigned short (*Bs)[LDP],
    const float* __restrict__ A, const unsigned short* __restrict__ Bt,
    int M, int N, int K, int m0, int n0, int k0, int tid)
{
  // ---- A: 128x32 f32 -> bf16, packed pairs, 1 ds_store_b32 each ----
#pragma unroll
  for (int i = 0; i < (BM * BK / 2) / 256; ++i) {   // 8 iters
    int e  = tid + i * 256;
    int r  = e >> 4, p = e & 15;                    // row, k-pair
    int gm = m0 + r;
    unsigned int pk = 0u;
    if (gm < M) {
      const float2 a = *(const float2*)(A + (size_t)gm * K + k0 + p * 2);
      pk = ((unsigned)f32_to_bf16(a.y) << 16) | (unsigned)f32_to_bf16(a.x);
      if (i == 0 && (k0 + BK) < K)                  // global_prefetch_b8
        __builtin_prefetch(A + (size_t)gm * K + k0 + BK, 0, 0);
    }
    *(unsigned int*)&As[r][p * 2] = pk;
  }
  // ---- B: 128 rows x 32 u16 (already [N][K]): 16B chunk copies ----
#pragma unroll
  for (int i = 0; i < (BN * 4) / 256; ++i) {        // 2 iters
    int e  = tid + i * 256;
    int c  = e >> 2, q = e & 3;                     // row, 8-u16 chunk
    int gn = n0 + c;
    v8us chunk = {0, 0, 0, 0, 0, 0, 0, 0};
    if (gn < N) chunk = *(const v8us*)(Bt + (size_t)gn * K + k0 + q * 8);
    *(v8us*)&Bs[c][q * 8] = chunk;
  }
}

__global__ __launch_bounds__(256)
void wmma_gemm_bias(const float* __restrict__ A,
                    const unsigned short* __restrict__ Bt,
                    const float* __restrict__ bias,
                    float* __restrict__ C,
                    int M, int N, int K, int relu)
{
  __shared__ __align__(16) unsigned short As[2][BM][LDP];  // 2 x 10 KB
  __shared__ __align__(16) unsigned short Bs[2][BN][LDP];  // 2 x 10 KB

  const int tid  = threadIdx.x;
  const int lane = tid & 31;
  const int wave = tid >> 5;      // 0..7
  const int wm   = wave & 3;      // wave's M strip (32 rows)
  const int wn   = wave >> 2;     // wave's N strip (64 cols)
  const int lh   = lane & 15;     // 0..15
  const int hi   = lane >> 4;     // 0/1 half-wave select
  const int m0   = blockIdx.y * BM;
  const int n0   = blockIdx.x * BN;

  FragC acc[2][4] = {};

  stage_tiles(As[0], Bs[0], A, Bt, M, N, K, m0, n0, 0, tid);
  int buf = 0;

  for (int k0 = 0; k0 < K; k0 += BK) {
    __syncthreads();
    if (k0 + BK < K)   // overlap next-tile global loads with this tile's WMMAs
      stage_tiles(As[buf ^ 1], Bs[buf ^ 1], A, Bt, M, N, K, m0, n0, k0 + BK, tid);

    // ---- load ALL fragments first (12x ds_load_b128), then WMMA burst ----
    FragAB af[2];
#pragma unroll
    for (int mi = 0; mi < 2; ++mi) {
      const unsigned short* arow = As[buf][wm * 32 + mi * 16 + lh];
      af[mi].h[0] = *(const v8us*)&arow[hi * 8];
      af[mi].h[1] = *(const v8us*)&arow[16 + hi * 8];
    }
    FragAB bfrag[4];
#pragma unroll
    for (int nb = 0; nb < 4; ++nb) {
      const unsigned short* brow = Bs[buf][wn * 64 + nb * 16 + lh];
      bfrag[nb].h[0] = *(const v8us*)&brow[hi * 8];
      bfrag[nb].h[1] = *(const v8us*)&brow[16 + hi * 8];
    }
    // 8 independent WMMAs (distinct accumulators -> no D->A/B hazard)
#pragma unroll
    for (int mi = 0; mi < 2; ++mi)
#pragma unroll
      for (int nb = 0; nb < 4; ++nb)
        acc[mi][nb].v = __builtin_amdgcn_wmma_f32_16x16x32_bf16(
            false, af[mi].v, false, bfrag[nb].v, (short)0, acc[mi][nb].v,
            false, false);
    buf ^= 1;
  }

  // ---- epilogue: f32 C/D layout ----
#pragma unroll
  for (int mi = 0; mi < 2; ++mi) {
#pragma unroll
    for (int nb = 0; nb < 4; ++nb) {
      int gn = n0 + wn * 64 + nb * 16 + lh;
      if (gn >= N) continue;
      float bv = bias ? bias[gn] : 0.f;
#pragma unroll
      for (int r = 0; r < 8; ++r) {
        int gm = m0 + wm * 32 + mi * 16 + hi * 8 + r;
        if (gm < M) {
          float v = acc[mi][nb].f[r] + bv;
          if (relu) v = fmaxf(v, 0.f);
          C[(size_t)gm * N + gn] = v;
        }
      }
    }
  }
}

// ---------------------------------------------------------------------------
// Weight conversion: f32 [K,N] -> bf16 transposed [N,K]
// ---------------------------------------------------------------------------
__global__ void cvt_bf16_t_kernel(const float* __restrict__ in,
                                  unsigned short* __restrict__ out,
                                  int K, int N) {
  int idx = blockIdx.x * blockDim.x + threadIdx.x;
  if (idx >= K * N) return;
  int k = idx / N, n = idx % N;          // coalesced read along n
  out[(size_t)n * K + k] = f32_to_bf16(in[idx]);
}

// ---------------------------------------------------------------------------
// Elementwise / graph kernels
// ---------------------------------------------------------------------------
__global__ void fill_f32_kernel(float* p, float v, int n) {
  int i = blockIdx.x * blockDim.x + threadIdx.x;
  if (i < n) p[i] = v;
}

__global__ void edge_deg_kernel(const int* __restrict__ dst,
                                const float* __restrict__ ew,
                                float* __restrict__ deg, int e) {
  int i = blockIdx.x * blockDim.x + threadIdx.x;
  if (i >= e) return;
  float w = ew ? ew[i] : 1.f;
  if (w != 0.f) atomicAdd(&deg[dst[i]], w);
}

__global__ void selfloop_kernel(const float* __restrict__ lin,
                                const float* __restrict__ deg,
                                const float* __restrict__ bias,
                                float* __restrict__ h, int n) {
  int idx = blockIdx.x * blockDim.x + threadIdx.x;
  if (idx >= n * HD) return;
  int r = idx / HD, c = idx % HD;
  h[idx] = lin[idx] / deg[r] + bias[c];
}

__global__ void edge_scatter_kernel(const int* __restrict__ src,
                                    const int* __restrict__ dst,
                                    const float* __restrict__ ew,
                                    const float* __restrict__ deg,
                                    const float* __restrict__ lin,
                                    float* __restrict__ h) {
  int e = blockIdx.x;
  int s = src[e], d = dst[e];
  float w = ew ? ew[e] : 1.f;
  if (w == 0.f) return;
  float nrm = w * rsqrtf(deg[s]) * rsqrtf(deg[d]);
  const float* ls = lin + (size_t)s * HD;
  float* hd = h + (size_t)d * HD;
  for (int c = threadIdx.x; c < HD; c += blockDim.x)
    atomicAdd(&hd[c], ls[c] * nrm);
}

__global__ void bn_stats_kernel(const float* __restrict__ h, int n,
                                float* __restrict__ mean,
                                float* __restrict__ rstd) {
  __shared__ float s1[256], s2[256];
  int c = blockIdx.x, t = threadIdx.x;
  float a = 0.f, b = 0.f;
  for (int r = t; r < n; r += 256) {
    float v = h[(size_t)r * HD + c];
    a += v; b += v * v;
  }
  s1[t] = a; s2[t] = b; __syncthreads();
  for (int o = 128; o > 0; o >>= 1) {
    if (t < o) { s1[t] += s1[t + o]; s2[t] += s2[t + o]; }
    __syncthreads();
  }
  if (t == 0) {
    float m = s1[0] / n;
    float var = s2[0] / n - m * m;   // biased var, as reference
    mean[c] = m;
    rstd[c] = rsqrtf(var + EPSBN);
  }
}

__global__ void bn_relu_kernel(float* __restrict__ h,
                               const float* __restrict__ mean,
                               const float* __restrict__ rstd,
                               const float* __restrict__ g,
                               const float* __restrict__ be, int n) {
  int idx = blockIdx.x * blockDim.x + threadIdx.x;
  if (idx >= n * HD) return;
  int c = idx % HD;
  float v = (h[idx] - mean[c]) * rstd[c] * g[c] + be[c];
  h[idx] = fmaxf(v, 0.f);
}

__global__ void pnorm_kernel(const float* __restrict__ p, float* __restrict__ out) {
  __shared__ float s[256];
  int t = threadIdx.x;
  float a = 0.f;
  for (int i = t; i < HD; i += 256) a += p[i] * p[i];
  s[t] = a; __syncthreads();
  for (int o = 128; o > 0; o >>= 1) {
    if (t < o) s[t] += s[t + o];
    __syncthreads();
  }
  if (t == 0) out[0] = rsqrtf(s[0]);
}

__global__ void score_kernel(const float* __restrict__ h,
                             const float* __restrict__ p,
                             const float* __restrict__ pinv,
                             float* __restrict__ score) {
  __shared__ float s[256];
  int row = blockIdx.x, t = threadIdx.x;
  const float* hr = h + (size_t)row * HD;
  float a = 0.f;
  for (int c = t; c < HD; c += 256) a += hr[c] * p[c];
  s[t] = a; __syncthreads();
  for (int o = 128; o > 0; o >>= 1) {
    if (t < o) s[t] += s[t + o];
    __syncthreads();
  }
  if (t == 0) score[row] = tanhf(s[0] * pinv[0]);
}

// per-graph top-k via bitonic sort in LDS; blockDim = n/2
__global__ void topk_kernel(const float* __restrict__ score, int n, int k,
                            int* __restrict__ perm, float* __restrict__ vals,
                            int* __restrict__ newid) {
  __shared__ float s[1024];
  __shared__ int   si[1024];
  int b = blockIdx.x, t = threadIdx.x, half = blockDim.x;
  for (int i = t; i < n; i += half) {
    s[i]  = score[b * n + i];
    si[i] = i;
    newid[b * n + i] = -1;
  }
  __syncthreads();
  for (int ksz = 2; ksz <= n; ksz <<= 1) {
    for (int j = ksz >> 1; j > 0; j >>= 1) {
      for (int i = t; i < n; i += half) {
        int ixj = i ^ j;
        if (ixj > i) {
          bool desc = ((i & ksz) == 0);
          bool sw = desc ? (s[i] < s[ixj]) : (s[i] > s[ixj]);
          if (sw) {
            float tv = s[i]; s[i] = s[ixj]; s[ixj] = tv;
            int   ti = si[i]; si[i] = si[ixj]; si[ixj] = ti;
          }
        }
      }
      __syncthreads();
    }
  }
  for (int i = t; i < k; i += half) {
    int idx = si[i];
    perm[b * k + i] = b * n + idx;
    vals[b * k + i] = s[i];
    newid[b * n + idx] = b * k + i;
  }
}

__global__ void gather_scale_kernel(const float* __restrict__ h,
                                    const int* __restrict__ perm,
                                    const float* __restrict__ vals,
                                    float* __restrict__ out, int total) {
  int idx = blockIdx.x * blockDim.x + threadIdx.x;
  if (idx >= total) return;
  int j = idx / HD, c = idx % HD;
  out[idx] = h[(size_t)perm[j] * HD + c] * vals[j];
}

__global__ void relabel_kernel(const int* __restrict__ src,
                               const int* __restrict__ dst,
                               const float* __restrict__ ew_in,
                               const int* __restrict__ newid,
                               int* __restrict__ src_o, int* __restrict__ dst_o,
                               float* __restrict__ ew_o, int e) {
  int i = blockIdx.x * blockDim.x + threadIdx.x;
  if (i >= e) return;
  int s = newid[src[i]], d = newid[dst[i]];
  bool v = (s >= 0) && (d >= 0);
  src_o[i] = v ? s : 0;
  dst_o[i] = v ? d : 0;
  ew_o[i]  = (ew_in ? ew_in[i] : 1.f) * (v ? 1.f : 0.f);
}

__global__ void mean_pool_kernel(const float* __restrict__ hp,
                                 float* __restrict__ x12, int k, int add) {
  int idx = blockIdx.x * blockDim.x + threadIdx.x;
  if (idx >= BG * HD) return;
  int b = idx / HD, c = idx % HD;
  float sum = 0.f;
  const float* base = hp + (size_t)b * k * HD + c;
  for (int j = 0; j < k; ++j) sum += base[(size_t)j * HD];
  float m = sum / (float)k;
  x12[idx] = add ? (x12[idx] + m) : m;
}

// ---------------------------------------------------------------------------
// Host-side launcher
// ---------------------------------------------------------------------------
static inline int ceil_div(int a, int b) { return (a + b - 1) / b; }

extern "C" void kernel_launch(void* const* d_in, const int* in_sizes, int n_in,
                              void* d_out, int out_size, void* d_ws, size_t ws_size,
                              hipStream_t stream) {
  (void)in_sizes; (void)n_in; (void)out_size; (void)ws_size;
  const float* x    = (const float*)d_in[0];
  const int*   eidx = (const int*)  d_in[1];
  /* batch d_in[2] unused (layout is contiguous per graph) */
  const float* W1   = (const float*)d_in[3];
  const float* b1   = (const float*)d_in[4];
  const float* g1   = (const float*)d_in[5];
  const float* be1  = (const float*)d_in[6];
  const float* p1   = (const float*)d_in[7];
  const float* W2   = (const float*)d_in[8];
  const float* b2   = (const float*)d_in[9];
  const float* g2   = (const float*)d_in[10];
  const float* be2  = (const float*)d_in[11];
  const float* p2   = (const float*)d_in[12];
  const float* Wf   = (const float*)d_in[13];
  const float* bf   = (const float*)d_in[14];
  const float* Wf1  = (const float*)d_in[15];
  const float* bf1  = (const float*)d_in[16];
  float* out = (float*)d_out;

  const int* src1 = eidx;        // edge_index[0]
  const int* dst1 = eidx + EE;   // edge_index[1]

  // ---- carve workspace ----
  char* w = (char*)d_ws;
  auto alloc = [&](size_t bytes) {
    void* p = (void*)w;
    w += (bytes + 255) & ~(size_t)255;
    return p;
  };
  float* lin   = (float*)alloc((size_t)N1 * HD * 4);
  float* hbuf  = (float*)alloc((size_t)N1 * HD * 4);
  float* hp    = (float*)alloc((size_t)N2 * HD * 4);
  float* deg   = (float*)alloc((size_t)N1 * 4);
  float* score = (float*)alloc((size_t)N1 * 4);
  float* vals  = (float*)alloc((size_t)N2 * 4);
  float* ew2   = (float*)alloc((size_t)EE * 4);
  float* x12   = (float*)alloc((size_t)BG * HD * 4);
  float* zbuf  = (float*)alloc((size_t)BG * 512 * 4);
  float* meanb = (float*)alloc((size_t)HD * 4);
  float* rstdb = (float*)alloc((size_t)HD * 4);
  float* pinv  = (float*)alloc(256);
  int*   perm  = (int*)alloc((size_t)N2 * 4);
  int*   newid = (int*)alloc((size_t)N1 * 4);
  int*   src2  = (int*)alloc((size_t)EE * 4);
  int*   dst2  = (int*)alloc((size_t)EE * 4);
  unsigned short* W1b  = (unsigned short*)alloc((size_t)F_IN * HD * 2);  // [HD][F_IN]
  unsigned short* W2b  = (unsigned short*)alloc((size_t)HD * HD * 2);    // [HD][HD]
  unsigned short* Wfb  = (unsigned short*)alloc((size_t)HD * 512 * 2);   // [512][HD]
  unsigned short* Wf1b = (unsigned short*)alloc((size_t)512 * OUTC * 2); // [OUTC][512]

  const int T = 256;

  // ---- weights -> bf16, transposed to [N][K] for contiguous B staging ----
  cvt_bf16_t_kernel<<<ceil_div(F_IN * HD, T), T, 0, stream>>>(W1, W1b, F_IN, HD);
  cvt_bf16_t_kernel<<<ceil_div(HD * HD, T), T, 0, stream>>>(W2, W2b, HD, HD);
  cvt_bf16_t_kernel<<<ceil_div(HD * 512, T), T, 0, stream>>>(Wf, Wfb, HD, 512);
  cvt_bf16_t_kernel<<<ceil_div(512 * OUTC, T), T, 0, stream>>>(Wf1, Wf1b, 512, OUTC);

  // ================= Layer 1: GCNConv + BN + ReLU =================
  wmma_gemm_bias<<<dim3(ceil_div(HD, BN), ceil_div(N1, BM)), 256, 0, stream>>>(
      x, W1b, nullptr, lin, N1, HD, F_IN, 0);
  fill_f32_kernel<<<ceil_div(N1, T), T, 0, stream>>>(deg, 1.f, N1);
  edge_deg_kernel<<<ceil_div(EE, T), T, 0, stream>>>(dst1, nullptr, deg, EE);
  selfloop_kernel<<<ceil_div(N1 * HD, T), T, 0, stream>>>(lin, deg, b1, hbuf, N1);
  edge_scatter_kernel<<<EE, T, 0, stream>>>(src1, dst1, nullptr, deg, lin, hbuf);
  bn_stats_kernel<<<HD, 256, 0, stream>>>(hbuf, N1, meanb, rstdb);
  bn_relu_kernel<<<ceil_div(N1 * HD, T), T, 0, stream>>>(hbuf, meanb, rstdb, g1, be1, N1);

  // ================= TopK pool 1 =================
  pnorm_kernel<<<1, 256, 0, stream>>>(p1, pinv);
  score_kernel<<<N1, 256, 0, stream>>>(hbuf, p1, pinv, score);
  topk_kernel<<<BG, NN / 2, 0, stream>>>(score, NN, KP1, perm, vals, newid);
  gather_scale_kernel<<<ceil_div(N2 * HD, T), T, 0, stream>>>(hbuf, perm, vals, hp, N2 * HD);
  relabel_kernel<<<ceil_div(EE, T), T, 0, stream>>>(src1, dst1, nullptr, newid,
                                                    src2, dst2, ew2, EE);
  mean_pool_kernel<<<ceil_div(BG * HD, T), T, 0, stream>>>(hp, x12, KP1, 0);

  // ================= Layer 2: GCNConv + BN + ReLU =================
  wmma_gemm_bias<<<dim3(ceil_div(HD, BN), ceil_div(N2, BM)), 256, 0, stream>>>(
      hp, W2b, nullptr, lin, N2, HD, HD, 0);
  fill_f32_kernel<<<ceil_div(N2, T), T, 0, stream>>>(deg, 1.f, N2);
  edge_deg_kernel<<<ceil_div(EE, T), T, 0, stream>>>(dst2, ew2, deg, EE);
  selfloop_kernel<<<ceil_div(N2 * HD, T), T, 0, stream>>>(lin, deg, b2, hbuf, N2);
  edge_scatter_kernel<<<EE, T, 0, stream>>>(src2, dst2, ew2, deg, lin, hbuf);
  bn_stats_kernel<<<HD, 256, 0, stream>>>(hbuf, N2, meanb, rstdb);
  bn_relu_kernel<<<ceil_div(N2 * HD, T), T, 0, stream>>>(hbuf, meanb, rstdb, g2, be2, N2);

  // ================= TopK pool 2 =================
  pnorm_kernel<<<1, 256, 0, stream>>>(p2, pinv);
  score_kernel<<<N2, 256, 0, stream>>>(hbuf, p2, pinv, score);
  topk_kernel<<<BG, KP1 / 2, 0, stream>>>(score, KP1, KP2, perm, vals, newid);
  gather_scale_kernel<<<ceil_div(N3 * HD, T), T, 0, stream>>>(hbuf, perm, vals, hp, N3 * HD);
  mean_pool_kernel<<<ceil_div(BG * HD, T), T, 0, stream>>>(hp, x12, KP2, 1);

  // ================= MLP head =================
  wmma_gemm_bias<<<dim3(ceil_div(512, BN), 1), 256, 0, stream>>>(
      x12, Wfb, bf, zbuf, BG, 512, HD, 1);
  wmma_gemm_bias<<<dim3(1, 1), 256, 0, stream>>>(
      zbuf, Wf1b, bf1, out, BG, OUTC, 512, 0);
}
